// DecoderLayer_1967095022014
// MI455X (gfx1250) — matrix operand order repaired and checked by reading, not compiled, not run on hardware
//
#include <hip/hip_runtime.h>
#include <math.h>

typedef unsigned short u16;
typedef unsigned int   u32;
typedef __attribute__((ext_vector_type(16))) __bf16 v16bf;
typedef __attribute__((ext_vector_type(8)))  float  v8f;
typedef __attribute__((ext_vector_type(4)))  int    v4i;

#define D_MODEL 2048
#define TD      6144      // 3 * D_MODEL
#define INNER   8192
#define SEQ     2048
#define HD      128       // head dim
#define NHEADS  16

// ---------- CDNA5 async global->LDS path (guarded; falls back to sync copy) ----
#if defined(__AMDGCN__) && __has_builtin(__builtin_amdgcn_global_load_async_to_lds_b128)
#define USE_ASYNC 1
#else
#define USE_ASYNC 0
#endif

__device__ __forceinline__ void async_copy_b128(const u16* g, u16* l) {
#if USE_ASYNC
    __builtin_amdgcn_global_load_async_to_lds_b128(
        (__attribute__((address_space(1))) v4i*)g,
        (__attribute__((address_space(3))) v4i*)l, 0, 0);
#else
    *(uint4*)l = *(const uint4*)g;
#endif
}
template<int N> __device__ __forceinline__ void wait_async() {
#if USE_ASYNC
#if __has_builtin(__builtin_amdgcn_s_wait_asynccnt)
    __builtin_amdgcn_s_wait_asynccnt(N);
#else
    asm volatile("s_wait_asynccnt %0" :: "n"(N) : "memory");
#endif
#endif
}

// ---------- helpers ----------
__device__ __forceinline__ u16 f2bf(float f) {
    union { float f; u32 u; } a; a.f = f;
    u32 u = a.u;
    u32 r = u + 0x7FFFu + ((u >> 16) & 1u);   // round-to-nearest-even
    return (u16)(r >> 16);
}

union FragBF { u32 u[8]; v16bf v; };

// ---------- f32 -> bf16 conversion (8 elems / thread) ----------
__global__ __launch_bounds__(256) void conv_bf16_kernel(const float* __restrict__ in,
                                                        u16* __restrict__ out, int n) {
    int i = (blockIdx.x * 256 + threadIdx.x) * 8;
    if (i + 7 >= n) return;
    float4 a = *(const float4*)(in + i);
    float4 b = *(const float4*)(in + i + 4);
    uint4 o;
    o.x = (u32)f2bf(a.x) | ((u32)f2bf(a.y) << 16);
    o.y = (u32)f2bf(a.z) | ((u32)f2bf(a.w) << 16);
    o.z = (u32)f2bf(b.x) | ((u32)f2bf(b.y) << 16);
    o.w = (u32)f2bf(b.z) | ((u32)f2bf(b.w) << 16);
    *(uint4*)(out + i) = o;
}

// ---------- f32 [K,N] -> bf16 [N,K] transpose+convert (32x32 LDS tiles) ----------
__global__ __launch_bounds__(256) void convT_bf16_kernel(const float* __restrict__ in,
                                                         u16* __restrict__ out,
                                                         int K, int N) {
    __shared__ u16 t[32][33];
    const int k0 = blockIdx.y * 32, n0 = blockIdx.x * 32;
    const int tid = threadIdx.x;
    const int r  = tid >> 3;
    const int c4 = (tid & 7) * 4;
    float4 v = *(const float4*)(in + (size_t)(k0 + r) * N + n0 + c4);
    t[r][c4 + 0] = f2bf(v.x);
    t[r][c4 + 1] = f2bf(v.y);
    t[r][c4 + 2] = f2bf(v.z);
    t[r][c4 + 3] = f2bf(v.w);
    __syncthreads();
    u16 a0 = t[c4 + 0][r], a1 = t[c4 + 1][r], a2 = t[c4 + 2][r], a3 = t[c4 + 3][r];
    uint2 o;
    o.x = (u32)a0 | ((u32)a1 << 16);
    o.y = (u32)a2 | ((u32)a3 << 16);
    *(uint2*)(out + (size_t)(n0 + r) * K + k0 + c4) = o;
}

// ---------- bf16 WMMA GEMM: C[M,N] = A[M,K] @ Bt[N,K]^T (+bias [+gelu|+resid]) ----
// A row-major [M,K]; Bt pre-transposed weights [N,K]. Double-buffered async LDS.
// EPI 0: out bf16 = acc + bias        (optional vT: V-projection columns scattered
//                                      into vT[col-2D][row] for attention)
// EPI 1: out bf16 = gelu(acc + bias)
// EPI 2: out f32  = acc + bias + resid
template<int EPI>
__global__ __launch_bounds__(256) void gemm_bf16(const u16* __restrict__ A,
                                                 const u16* __restrict__ Bt,
                                                 const float* __restrict__ bias,
                                                 const float* __restrict__ resid,
                                                 u16* __restrict__ vT,
                                                 void* __restrict__ outp,
                                                 int M, int N, int K) {
    __shared__ u16 Al[2][128 * 40];   // [m][k], stride 40
    __shared__ u16 Bl[2][128 * 40];   // [n][k], stride 40

    const int tid  = threadIdx.x;
    const int lane = tid & 31;
    const int wid  = tid >> 5;          // 0..7
    const int wm   = wid & 3;           // 4 wave rows  (32 M each)
    const int wn   = wid >> 2;          // 2 wave cols  (64 N each)
    const int half = lane >> 4;
    const int l16  = lane & 15;
    const int m0   = blockIdx.y * 128;
    const int n0   = blockIdx.x * 128;

    v8f acc[2][4] = {};

    auto issue = [&](int k0, int buf) {
        #pragma unroll
        for (int p = 0; p < 2; p++) {
            int id = p * 256 + tid;
            int r = id >> 2, sg = id & 3;
            async_copy_b128(A  + (size_t)(m0 + r) * K + k0 + sg * 8, &Al[buf][r * 40 + sg * 8]);
            async_copy_b128(Bt + (size_t)(n0 + r) * K + k0 + sg * 8, &Bl[buf][r * 40 + sg * 8]);
        }
    };

    const int nk = K >> 5;
    issue(0, 0);
    for (int t = 0; t < nk; t++) {
        const int cur = t & 1;
        if (t + 1 < nk) { issue((t + 1) << 5, (t + 1) & 1); wait_async<4>(); }
        else            { wait_async<0>(); }
        __syncthreads();

        FragBF af[2], bf[4];
        #pragma unroll
        for (int tm = 0; tm < 2; tm++) {
            int mr = wm * 32 + tm * 16 + l16;
            #pragma unroll
            for (int v = 0; v < 8; v++) {
                int kk = ((v >> 2) << 4) + (half << 3) + ((v & 3) << 1); // ISA 16-bit A layout
                af[tm].u[v] = *(const u32*)(&Al[cur][mr * 40 + kk]);
            }
        }
        #pragma unroll
        for (int tn = 0; tn < 4; tn++) {
            int nc = wn * 64 + tn * 16 + l16;
            #pragma unroll
            for (int v = 0; v < 8; v++) {
                int kk = ((v >> 2) << 4) + (half << 3) + ((v & 3) << 1); // ISA 16-bit B layout
                bf[tn].u[v] = *(const u32*)(&Bl[cur][nc * 40 + kk]);
            }
        }
        #pragma unroll
        for (int tm = 0; tm < 2; tm++)
            #pragma unroll
            for (int tn = 0; tn < 4; tn++)
                acc[tm][tn] = __builtin_amdgcn_wmma_f32_16x16x32_bf16(
                    false, af[tm].v, false, bf[tn].v, (short)0, acc[tm][tn], false, false);
        __syncthreads();
    }

    // epilogue
    #pragma unroll
    for (int tn = 0; tn < 4; tn++) {
        int col = n0 + wn * 64 + tn * 16 + l16;
        float bv = bias ? bias[col] : 0.0f;
        #pragma unroll
        for (int tm = 0; tm < 2; tm++) {
            #pragma unroll
            for (int v = 0; v < 8; v++) {
                int row = m0 + wm * 32 + tm * 16 + v + half * 8;  // ISA C layout
                float val = acc[tm][tn][v] + bv;
                size_t o = (size_t)row * N + col;
                if constexpr (EPI == 1)
                    val = 0.5f * val * (1.0f + erff(val * 0.70710678118654752f));
                if constexpr (EPI == 2) {
                    ((float*)outp)[o] = val + resid[o];
                } else {
                    if (EPI == 0 && vT != nullptr && col >= 2 * D_MODEL)
                        vT[(size_t)(col - 2 * D_MODEL) * SEQ + row] = f2bf(val);
                    else
                        ((u16*)outp)[o] = f2bf(val);
                }
            }
        }
    }
}

// ---------- flash attention (causal), bf16 WMMA, online softmax ----------
// qkv: [SEQ, 3*D_MODEL] bf16 (Q | K | unused-V), vT: [D_MODEL, SEQ] bf16 (V^T)
// ctx: [SEQ, D_MODEL] bf16
__global__ __launch_bounds__(256) void attn_kernel(const u16* __restrict__ qkv,
                                                   const u16* __restrict__ vT,
                                                   u16* __restrict__ ctx) {
    const int head = blockIdx.x;
    const int q0   = blockIdx.y * 128;
    const int tid  = threadIdx.x;
    const int lane = tid & 31;
    const int wid  = tid >> 5;
    const int half = lane >> 4;
    const int l16  = lane & 15;

    __shared__ u16 Kl[2][32 * 136];     // [key][d]   (K-contig for S B-frag)
    __shared__ u16 Vl[2][128 * 40];     // [d][key]   (key-contig for PV B-frag)
    __shared__ u16 Pl[8 * 16 * 40];     // per-wave P tile [m][key]

    // Q fragments for this wave's 16 rows, 4 K-chunks of 32 along head dim
    FragBF qf[4];
    {
        const int qrow = q0 + wid * 16 + l16;
        const u32* qb = (const u32*)(qkv + (size_t)qrow * TD + head * HD);
        #pragma unroll
        for (int c = 0; c < 4; c++)
            #pragma unroll
            for (int v = 0; v < 8; v++) {
                int d = c * 32 + ((v >> 2) << 4) + (half << 3) + ((v & 3) << 1);
                qf[c].u[v] = qb[d >> 1];
            }
    }

    v8f oacc[8] = {};                   // O accumulator 16x128 (8 d-tiles)
    float mrun[8], lrun[8];
    #pragma unroll
    for (int v = 0; v < 8; v++) { mrun[v] = -1e30f; lrun[v] = 0.0f; }

    const float sc = 0.022097086912079608f;   // 1/sqrt(D_MODEL) (faithful to ref)
    const int nchunks = (q0 + 128) >> 5;      // causal: no chunks past q-block

    auto issue = [&](int ch, int buf) {
        const int k0 = ch << 5;
        #pragma unroll
        for (int p = 0; p < 2; p++) {
            int id = p * 256 + tid;
            int key = id >> 4, sg = id & 15;               // K tile: 32 keys x 128 d
            async_copy_b128(qkv + (size_t)(k0 + key) * TD + D_MODEL + head * HD + sg * 8,
                            &Kl[buf][key * 136 + sg * 8]);
            int dr = id >> 2, sg2 = id & 3;                // V^T tile: 128 d x 32 keys
            async_copy_b128(vT + (size_t)(head * HD + dr) * SEQ + k0 + sg2 * 8,
                            &Vl[buf][dr * 40 + sg2 * 8]);
        }
    };

    issue(0, 0);
    for (int ch = 0; ch < nchunks; ch++) {
        const int cur = ch & 1;
        const int k0  = ch << 5;
        if (ch + 1 < nchunks) { issue(ch + 1, (ch + 1) & 1); wait_async<4>(); }
        else                  { wait_async<0>(); }
        __syncthreads();

        // S = Q @ K^T  (16 x 32)
        v8f s[2] = {};
        #pragma unroll
        for (int tn = 0; tn < 2; tn++) {
            int key = tn * 16 + l16;
            #pragma unroll
            for (int c = 0; c < 4; c++) {
                FragBF bf;
                #pragma unroll
                for (int v = 0; v < 8; v++) {
                    int d = c * 32 + ((v >> 2) << 4) + (half << 3) + ((v & 3) << 1);
                    bf.u[v] = *(const u32*)(&Kl[cur][key * 136 + d]);
                }
                s[tn] = __builtin_amdgcn_wmma_f32_16x16x32_bf16(
                    false, qf[c].v, false, bf.v, (short)0, s[tn], false, false);
            }
        }

        // scale, causal mask, online softmax, emit P (bf16) to LDS
        #pragma unroll
        for (int v = 0; v < 8; v++) {
            int r = q0 + wid * 16 + v + half * 8;
            float p0 = s[0][v] * sc;
            float p1 = s[1][v] * sc;
            if (k0 + l16 > r)      p0 = -1e30f;
            if (k0 + 16 + l16 > r) p1 = -1e30f;
            float m = fmaxf(p0, p1);
            #pragma unroll
            for (int off = 1; off < 16; off <<= 1) m = fmaxf(m, __shfl_xor(m, off, 32));
            float mnew  = fmaxf(mrun[v], m);
            float alpha = __expf(mrun[v] - mnew);
            mrun[v] = mnew;
            float e0 = __expf(p0 - mnew);
            float e1 = __expf(p1 - mnew);
            float rs = e0 + e1;
            #pragma unroll
            for (int off = 1; off < 16; off <<= 1) rs += __shfl_xor(rs, off, 32);
            lrun[v] = lrun[v] * alpha + rs;
            #pragma unroll
            for (int dt = 0; dt < 8; dt++) oacc[dt][v] *= alpha;
            int prow = wid * 16 + v + half * 8;
            Pl[prow * 40 + l16]      = f2bf(e0);
            Pl[prow * 40 + 16 + l16] = f2bf(e1);
        }
        __syncthreads();

        // O += P @ V
        FragBF pa;
        #pragma unroll
        for (int v = 0; v < 8; v++) {
            int kk = ((v >> 2) << 4) + (half << 3) + ((v & 3) << 1);
            pa.u[v] = *(const u32*)(&Pl[(wid * 16 + l16) * 40 + kk]);
        }
        #pragma unroll
        for (int dt = 0; dt < 8; dt++) {
            FragBF vb;
            int n = dt * 16 + l16;
            #pragma unroll
            for (int v = 0; v < 8; v++) {
                int kk = ((v >> 2) << 4) + (half << 3) + ((v & 3) << 1);
                vb.u[v] = *(const u32*)(&Vl[cur][n * 40 + kk]);
            }
            oacc[dt] = __builtin_amdgcn_wmma_f32_16x16x32_bf16(
                false, pa.v, false, vb.v, (short)0, oacc[dt], false, false);
        }
        __syncthreads();   // protects Pl and LDS buffers for next iteration
    }

    // normalize and store context (bf16)
    #pragma unroll
    for (int v = 0; v < 8; v++) {
        float inv = 1.0f / lrun[v];
        int row = q0 + wid * 16 + v + half * 8;
        #pragma unroll
        for (int dt = 0; dt < 8; dt++) {
            int col = head * HD + dt * 16 + l16;
            ctx[(size_t)row * D_MODEL + col] = f2bf(oacc[dt][v] * inv);
        }
    }
}

// ---------- LayerNorm: one block per row of D_MODEL ----------
__global__ __launch_bounds__(256) void ln_kernel(const float* __restrict__ in,
                                                 const float* __restrict__ g,
                                                 const float* __restrict__ b,
                                                 float* __restrict__ outf,
                                                 u16* __restrict__ outb) {
    const int row = blockIdx.x, tid = threadIdx.x;
    const float* x = in + (size_t)row * D_MODEL;
    float vbuf[8], s = 0.f, s2 = 0.f;
    #pragma unroll
    for (int i = 0; i < 8; i++) {
        float t = x[tid + i * 256];
        vbuf[i] = t; s += t; s2 += t * t;
    }
    __shared__ float red[512];
    red[tid] = s; red[256 + tid] = s2;
    __syncthreads();
    for (int off = 128; off > 0; off >>= 1) {
        if (tid < off) { red[tid] += red[tid + off]; red[256 + tid] += red[256 + tid + off]; }
        __syncthreads();
    }
    float mu  = red[0] * (1.0f / D_MODEL);
    float var = red[256] * (1.0f / D_MODEL) - mu * mu;
    float rs  = rsqrtf(var + 1e-5f);
    #pragma unroll
    for (int i = 0; i < 8; i++) {
        int c = tid + i * 256;
        float y = (vbuf[i] - mu) * rs * g[c] + b[c];
        outf[(size_t)row * D_MODEL + c] = y;
        if (outb) outb[(size_t)row * D_MODEL + c] = f2bf(y);
    }
}

// ---------- launcher ----------
extern "C" void kernel_launch(void* const* d_in, const int* in_sizes, int n_in,
                              void* d_out, int out_size, void* d_ws, size_t ws_size,
                              hipStream_t stream) {
    const float* x    = (const float*)d_in[0];
    const float* Wqkv = (const float*)d_in[1];
    const float* bqkv = (const float*)d_in[2];
    const float* Wo   = (const float*)d_in[3];
    const float* bo   = (const float*)d_in[4];
    const float* W1   = (const float*)d_in[5];
    const float* b1   = (const float*)d_in[6];
    const float* W2   = (const float*)d_in[7];
    const float* b2   = (const float*)d_in[8];
    const float* g1   = (const float*)d_in[9];
    const float* be1  = (const float*)d_in[10];
    const float* g2   = (const float*)d_in[11];
    const float* be2  = (const float*)d_in[12];
    float* out = (float*)d_out;

    // workspace partition
    char* w = (char*)d_ws;
    auto take = [&](size_t bytes) { char* p = w; w += (bytes + 255) & ~(size_t)255; return p; };
    u16*   xb     = (u16*)  take((size_t)SEQ * D_MODEL * 2);
    u16*   wqkvT  = (u16*)  take((size_t)TD * D_MODEL * 2);     // [TD, D]
    u16*   woT    = (u16*)  take((size_t)D_MODEL * D_MODEL * 2);// [D, D]
    u16*   w1T    = (u16*)  take((size_t)INNER * D_MODEL * 2);  // [INNER, D]
    u16*   w2T    = (u16*)  take((size_t)D_MODEL * INNER * 2);  // [D, INNER]
    u16*   qkv_b  = (u16*)  take((size_t)SEQ * TD * 2);
    u16*   vT     = (u16*)  take((size_t)D_MODEL * SEQ * 2);    // [D, SEQ]
    u16*   ctx_b  = (u16*)  take((size_t)SEQ * D_MODEL * 2);
    float* s1     = (float*)take((size_t)SEQ * D_MODEL * 4);
    float* h_f    = (float*)take((size_t)SEQ * D_MODEL * 4);
    u16*   h_b    = (u16*)  take((size_t)SEQ * D_MODEL * 2);
    u16*   ff1_b  = (u16*)  take((size_t)SEQ * INNER * 2);
    float* s2     = (float*)take((size_t)SEQ * D_MODEL * 4);

    // 1) bf16 conversions: x direct; weights transposed to [N,K]
    conv_bf16_kernel<<<dim3(SEQ * D_MODEL / 2048), dim3(256), 0, stream>>>(
        x, xb, SEQ * D_MODEL);
    convT_bf16_kernel<<<dim3(TD / 32, D_MODEL / 32), dim3(256), 0, stream>>>(
        Wqkv, wqkvT, D_MODEL, TD);
    convT_bf16_kernel<<<dim3(D_MODEL / 32, D_MODEL / 32), dim3(256), 0, stream>>>(
        Wo, woT, D_MODEL, D_MODEL);
    convT_bf16_kernel<<<dim3(INNER / 32, D_MODEL / 32), dim3(256), 0, stream>>>(
        W1, w1T, D_MODEL, INNER);
    convT_bf16_kernel<<<dim3(D_MODEL / 32, INNER / 32), dim3(256), 0, stream>>>(
        W2, w2T, INNER, D_MODEL);

    // 2) qkv = x @ Wqkv + bqkv -> bf16 (V columns scattered into vT)
    gemm_bf16<0><<<dim3(TD / 128, SEQ / 128), dim3(256), 0, stream>>>(
        xb, wqkvT, bqkv, nullptr, vT, qkv_b, SEQ, TD, D_MODEL);

    // 3) causal flash attention -> ctx bf16
    attn_kernel<<<dim3(NHEADS, SEQ / 128), dim3(256), 0, stream>>>(qkv_b, vT, ctx_b);

    // 4) s1 = ctx @ Wo + bo + x -> f32
    gemm_bf16<2><<<dim3(D_MODEL / 128, SEQ / 128), dim3(256), 0, stream>>>(
        ctx_b, woT, bo, x, nullptr, s1, SEQ, D_MODEL, D_MODEL);

    // 5) h = LN(s1) -> f32 + bf16
    ln_kernel<<<dim3(SEQ), dim3(256), 0, stream>>>(s1, g1, be1, h_f, h_b);

    // 6) ff1 = gelu(h @ W1 + b1) -> bf16
    gemm_bf16<1><<<dim3(INNER / 128, SEQ / 128), dim3(256), 0, stream>>>(
        h_b, w1T, b1, nullptr, nullptr, ff1_b, SEQ, INNER, D_MODEL);

    // 7) s2 = ff1 @ W2 + b2 + h -> f32
    gemm_bf16<2><<<dim3(D_MODEL / 128, SEQ / 128), dim3(256), 0, stream>>>(
        ff1_b, w2T, b2, h_f, nullptr, s2, SEQ, D_MODEL, INNER);

    // 8) out = LN(s2) -> f32
    ln_kernel<<<dim3(SEQ), dim3(256), 0, stream>>>(s2, g2, be2, out, nullptr);

    (void)in_sizes; (void)n_in; (void)out_size; (void)ws_size;
}
